// S4Layer_76802605187519
// MI455X (gfx1250) — matrix operand order repaired and checked by reading, not compiled, not run on hardware
//
#include <hip/hip_runtime.h>
#include <hip/hip_bf16.h>

// ---- problem dims (fixed by reference) ----
#define D_DIM 1024
#define N_DIM 128
#define L_DIM 4096
#define B_DIM 8
#define M_DIM (B_DIM * L_DIM)      // 32768 rows for both GEMMs
#define NCHUNK 128                 // scan chunks along L
#define CHUNK 32                   // L_DIM / NCHUNK
#define LROW 36                    // padded LDS row stride in f32 (144 B)

typedef __attribute__((ext_vector_type(16))) __bf16 v16bf;
typedef __attribute__((ext_vector_type(8)))  __bf16 v8bf;
typedef __attribute__((ext_vector_type(8)))  float  v8f;

union BF16Frag { v16bf v; v8bf h[2]; };

// 4 x async b128: copies 64 contiguous bytes global -> LDS (ASYNCcnt += 4).
// GVS mode: mem = SGPR64 + VGPR32 + inst_offset; LDS = VGPR(laddr) + inst_offset.
#define ASYNC_LOAD64B(laddr, voff, sbase)                                      \
    asm volatile("global_load_async_to_lds_b128 %0, %1, %2\n\t"                \
                 "global_load_async_to_lds_b128 %0, %1, %2 offset:16\n\t"      \
                 "global_load_async_to_lds_b128 %0, %1, %2 offset:32\n\t"      \
                 "global_load_async_to_lds_b128 %0, %1, %2 offset:48"          \
                 :: "v"(laddr), "v"(voff), "s"(sbase) : "memory")

__device__ __forceinline__ v8bf cvt8(float4 a, float4 b) {
    v8bf r;
    r[0] = (__bf16)a.x; r[1] = (__bf16)a.y; r[2] = (__bf16)a.z; r[3] = (__bf16)a.w;
    r[4] = (__bf16)b.x; r[5] = (__bf16)b.y; r[6] = (__bf16)b.z; r[7] = (__bf16)b.w;
    return r;
}

// ---------------------------------------------------------------------------
// Kernel 0: convert B_w (128x1024) and C_w (1024x128) to bf16 in workspace.
// ---------------------------------------------------------------------------
__global__ __launch_bounds__(256)
void s4_cvt_weights(const float* __restrict__ Bw_f, const float* __restrict__ Cw_f,
                    __bf16* __restrict__ Bwb, __bf16* __restrict__ Cwb) {
    int i = blockIdx.x * 256 + threadIdx.x;            // grid = 1024 -> 262144 ids
    if (i < N_DIM * D_DIM) {
        Bwb[i] = (__bf16)Bw_f[i];
    } else {
        int j = i - N_DIM * D_DIM;
        Cwb[j] = (__bf16)Cw_f[j];
    }
}

// ---------------------------------------------------------------------------
// Kernel 1: x_proj[m, n] = sum_d x[m, d] * B_w[n, d]   (M=32768, K=1024, N=128)
// 256 thr (8 waves), block tile 128(M) x 128(N).
// x tile (128x32 f32) DMA'd into double-buffered LDS via
// GLOBAL_LOAD_ASYNC_TO_LDS_B128 (next-tile copy overlaps WMMA on current
// tile); f32->bf16 conversion happens in the fragment build. A sched_barrier
// separates fragment loads from the 8 back-to-back WMMAs.
// ---------------------------------------------------------------------------
__global__ __launch_bounds__(256)
void s4_gemm1(const float* __restrict__ x, const __bf16* __restrict__ Bw,
              float* __restrict__ xp) {
    __shared__ float lx[2][128 * LROW];                // 2 x 18 KB, padded rows
    const int t    = threadIdx.x;
    const int wave = t >> 5;
    const int lane = t & 31;
    const int m0   = blockIdx.x * 128;
    const int half = lane >> 4;                        // 0: lanes 0-15, 1: lanes 16-31
    const int lrow = (wave << 4) + (lane & 15);        // LDS row for A fragment

    v8f acc[8] = {};

    // staging assignment: thread t copies row = t>>1, 64B segment seg = t&1
    const int ldr = t >> 1;
    const int seg = t & 1;
    const uint64_t xbase = (uint64_t)(uintptr_t)x;
    uint32_t gv = (uint32_t)(((m0 + ldr) * D_DIM + seg * 16) * 4); // byte offset
    const uint32_t l0 = (uint32_t)(uintptr_t)&lx[0][ldr * LROW + seg * 16];
    const uint32_t l1 = (uint32_t)(uintptr_t)&lx[1][ldr * LROW + seg * 16];

    // prologue: DMA tile 0 into buffer 0
    ASYNC_LOAD64B(l0, gv, xbase);

    for (int kk = 0; kk < D_DIM; kk += 32) {
        const int cur = (kk >> 5) & 1;
        const bool has_next = (kk + 32) < D_DIM;
        if (has_next) {
            ASYNC_LOAD64B(cur ? l0 : l1, gv + 128u, xbase);     // next tile
            asm volatile("s_wait_asynccnt 0x4" ::: "memory");   // cur tile landed
        } else {
            asm volatile("s_wait_asynccnt 0x0" ::: "memory");
        }
        __syncthreads();                                        // all waves' DMA visible

        // preload all 8 B fragments (L2-resident bf16 B_w):
        // col = nt*16 + (lane&15), K = kk + 16*half + e (16 contiguous bf16)
        BF16Frag bf[8];
        const __bf16* bp = Bw + (size_t)(lane & 15) * D_DIM + kk + half * 16;
        #pragma unroll
        for (int nt = 0; nt < 8; ++nt) {
            const __bf16* bpt = bp + (size_t)(nt * 16) * D_DIM;
            bf[nt].h[0] = *(const v8bf*)(bpt);
            bf[nt].h[1] = *(const v8bf*)(bpt + 8);
        }

        // A fragment from f32 LDS tile (contiguous K, matching B):
        // lanes 0-15 K=0-15 ; lanes 16-31 K=16-31
        const float* ap = &lx[cur][lrow * LROW + half * 16];
        float4 g0 = ((const float4*)ap)[0];
        float4 g1 = ((const float4*)ap)[1];
        float4 g2 = ((const float4*)(ap + 8))[0];
        float4 g3 = ((const float4*)(ap + 8))[1];
        BF16Frag a;
        a.h[0] = cvt8(g0, g1);
        a.h[1] = cvt8(g2, g3);

        // keep all fragment loads ahead of the WMMA burst
        __builtin_amdgcn_sched_barrier(0);

        #pragma unroll
        for (int nt = 0; nt < 8; ++nt)
            acc[nt] = __builtin_amdgcn_wmma_f32_16x16x32_bf16(
                false, a.v, false, bf[nt].v, (short)0, acc[nt], false, false);

        __syncthreads();      // reads of buffer `cur` done before it is re-filled
        gv += 128u;
    }

    // store x_proj (f32): C/D layout -> row = r + 8*half, col = lane&15
    const int rbase = half * 8;
    #pragma unroll
    for (int nt = 0; nt < 8; ++nt) {
        int ncol = nt * 16 + (lane & 15);
        #pragma unroll
        for (int r = 0; r < 8; ++r) {
            int m = m0 + (wave << 4) + rbase + r;
            xp[(size_t)m * N_DIM + ncol] = acc[nt][r];
        }
    }
}

// ---------------------------------------------------------------------------
// Kernel 2a: per-chunk carries of h = A*h + u
// ---------------------------------------------------------------------------
__global__ __launch_bounds__(128)
void s4_scan_carry(const float* __restrict__ xp, const float* __restrict__ logA,
                   float* __restrict__ carry) {
    const int b = blockIdx.x >> 7;
    const int c = blockIdx.x & (NCHUNK - 1);
    const int n = threadIdx.x;
    const float A = __expf(logA[n]);
    const float* p = xp + ((size_t)b * L_DIM + (size_t)c * CHUNK) * N_DIM + n;
    float h = 0.f;
    #pragma unroll 8
    for (int i = 0; i < CHUNK; ++i) h = fmaf(A, h, p[(size_t)i * N_DIM]);
    carry[((size_t)b * NCHUNK + c) * N_DIM + n] = h;
}

// ---------------------------------------------------------------------------
// Kernel 2b: exclusive combine of carries across chunks with factor A^CHUNK.
// ---------------------------------------------------------------------------
__global__ __launch_bounds__(128)
void s4_scan_mid(const float* __restrict__ logA, float* __restrict__ carry) {
    const int b = blockIdx.x;
    const int n = threadIdx.x;
    const float Ac = __expf(logA[n] * (float)CHUNK);
    float run = 0.f;
    for (int j = 0; j < NCHUNK; ++j) {
        size_t idx = ((size_t)b * NCHUNK + j) * N_DIM + n;
        float cj = carry[idx];
        carry[idx] = run;                  // carry-in (exclusive)
        run = fmaf(Ac, run, cj);
    }
}

// ---------------------------------------------------------------------------
// Kernel 2c: recompute chunk scans seeded with carry-in, emit hs in bf16.
// ---------------------------------------------------------------------------
__global__ __launch_bounds__(128)
void s4_scan_final(const float* __restrict__ xp, const float* __restrict__ logA,
                   const float* __restrict__ carry, __bf16* __restrict__ hs) {
    const int b = blockIdx.x >> 7;
    const int c = blockIdx.x & (NCHUNK - 1);
    const int n = threadIdx.x;
    const float A = __expf(logA[n]);
    float h = carry[((size_t)b * NCHUNK + c) * N_DIM + n];
    const float* p = xp + ((size_t)b * L_DIM + (size_t)c * CHUNK) * N_DIM + n;
    __bf16*      o = hs + ((size_t)b * L_DIM + (size_t)c * CHUNK) * N_DIM + n;
    #pragma unroll 8
    for (int i = 0; i < CHUNK; ++i) {
        h = fmaf(A, h, p[(size_t)i * N_DIM]);
        o[(size_t)i * N_DIM] = (__bf16)h;
    }
}

// ---------------------------------------------------------------------------
// Kernel 3: y[m, d] = sum_n hs[m, n] * C_w[d, n] + x[m, d] * D_param[d]
// K=128 fully unrolled (4 k-steps x 8 d-tiles = 32 WMMA); per k-step the A
// fragment and all 8 B fragments are preloaded (sched_barrier keeps them
// ahead of the WMMA burst). Epilogue: last-use x via nontemporal load,
// streaming y via nontemporal store (don't churn L2 with a write-once 134MB).
// ---------------------------------------------------------------------------
__global__ __launch_bounds__(256)
void s4_gemm2(const __bf16* __restrict__ hs, const __bf16* __restrict__ Cw,
              const float* __restrict__ x, const float* __restrict__ Dp,
              float* __restrict__ y) {
    const int t    = threadIdx.x;
    const int wave = t >> 5;
    const int lane = t & 31;
    const int m0   = blockIdx.x * 128;
    const int d0   = blockIdx.y * 128;
    const int half = lane >> 4;
    const int mrow = m0 + (wave << 4) + (lane & 15);

    v8f acc[8] = {};

    #pragma unroll
    for (int kk = 0; kk < N_DIM; kk += 32) {
        BF16Frag a, bf[8];
        const __bf16* ap = hs + (size_t)mrow * N_DIM + kk + half * 16;
        a.h[0] = *(const v8bf*)(ap);
        a.h[1] = *(const v8bf*)(ap + 8);
        #pragma unroll
        for (int nt = 0; nt < 8; ++nt) {
            const __bf16* bpt =
                Cw + (size_t)(d0 + nt * 16 + (lane & 15)) * N_DIM + kk + half * 16;
            bf[nt].h[0] = *(const v8bf*)(bpt);
            bf[nt].h[1] = *(const v8bf*)(bpt + 8);
        }
        __builtin_amdgcn_sched_barrier(0);
        #pragma unroll
        for (int nt = 0; nt < 8; ++nt)
            acc[nt] = __builtin_amdgcn_wmma_f32_16x16x32_bf16(
                false, a.v, false, bf[nt].v, (short)0, acc[nt], false, false);
    }

    // fused epilogue: + x * D_param, store f32 y (streaming)
    const int rbase = half * 8;
    #pragma unroll
    for (int nt = 0; nt < 8; ++nt) {
        int d = d0 + nt * 16 + (lane & 15);
        float dp = Dp[d];
        #pragma unroll
        for (int r = 0; r < 8; ++r) {
            size_t idx = (size_t)(m0 + (wave << 4) + rbase + r) * D_DIM + d;
            float xv = __builtin_nontemporal_load(&x[idx]);     // last use of x
            __builtin_nontemporal_store(acc[nt][r] + xv * dp, &y[idx]);
        }
    }
}

// ---------------------------------------------------------------------------
// Kernel 4: final_state = broadcast(mean_b x_proj[:, L-1, :]) -> (B, N)
// ---------------------------------------------------------------------------
__global__ __launch_bounds__(128)
void s4_final_state(const float* __restrict__ xp, float* __restrict__ out) {
    const int n = threadIdx.x;
    float s = 0.f;
    #pragma unroll
    for (int b = 0; b < B_DIM; ++b)
        s += xp[((size_t)b * L_DIM + (L_DIM - 1)) * N_DIM + n];
    s *= (1.0f / B_DIM);
    #pragma unroll
    for (int b = 0; b < B_DIM; ++b)
        out[(size_t)M_DIM * D_DIM + (size_t)b * N_DIM + n] = s;
}

// ---------------------------------------------------------------------------
extern "C" void kernel_launch(void* const* d_in, const int* in_sizes, int n_in,
                              void* d_out, int out_size, void* d_ws, size_t ws_size,
                              hipStream_t stream) {
    const float* x    = (const float*)d_in[0];   // (8, 4096, 1024)
    const float* B_w  = (const float*)d_in[1];   // (128, 1024)
    const float* C_w  = (const float*)d_in[2];   // (1024, 128)
    const float* logA = (const float*)d_in[3];   // (128,)
    const float* Dp   = (const float*)d_in[4];   // (1024,)
    float* out = (float*)d_out;                  // y (B,L,D) then final_state (B,N)

    // workspace layout
    char* ws = (char*)d_ws;
    __bf16* Bwb   = (__bf16*)(ws);                                      // 256 KB
    __bf16* Cwb   = (__bf16*)(ws + (size_t)262144);                     // 256 KB
    float*  xp    = (float*) (ws + (size_t)524288);                     // 16 MB
    float*  carry = (float*) (ws + (size_t)524288 + 16777216);          // 512 KB
    __bf16* hs    = (__bf16*)(ws + (size_t)524288 + 16777216 + 524288); // 8 MB

    s4_cvt_weights<<<1024, 256, 0, stream>>>(B_w, C_w, Bwb, Cwb);
    s4_gemm1<<<M_DIM / 128, 256, 0, stream>>>(x, Bwb, xp);
    s4_scan_carry<<<B_DIM * NCHUNK, 128, 0, stream>>>(xp, logA, carry);
    s4_scan_mid<<<B_DIM, 128, 0, stream>>>(logA, carry);
    s4_scan_final<<<B_DIM * NCHUNK, 128, 0, stream>>>(xp, logA, carry, hs);
    dim3 g2(M_DIM / 128, D_DIM / 128);
    s4_gemm2<<<g2, 256, 0, stream>>>(hs, Cwb, x, Dp, out);
    s4_final_state<<<1, 128, 0, stream>>>(xp, out);
}
